// MultiHeadAttention_31035433681664
// MI455X (gfx1250) — matrix operand order repaired and checked by reading, not compiled, not run on hardware
//
#include <hip/hip_runtime.h>

// ---------------------------------------------------------------------------
// Transformer-XL relative attention for MI455X (gfx1250, wave32, WMMA).
// All GEMMs use v_wmma_f32_16x16x32_bf16 (bf16 in, fp32 accumulate).
// ---------------------------------------------------------------------------

#define DEV __device__ __forceinline__

typedef __attribute__((ext_vector_type(16))) __bf16 v16bf;
typedef __attribute__((ext_vector_type(8)))  __bf16 v8bf;
typedef __attribute__((ext_vector_type(8)))  float  v8f;

constexpr int Bb  = 16;    // batch
constexpr int Ss  = 256;   // sequence
constexpr int Dd  = 1024;  // model dim
constexpr int Hh  = 16;    // heads
constexpr int DKk = 64;    // head dim
constexpr int GW  = 272;   // BD-strip columns (j = z - s + 255 in [0, 271])
constexpr int SSW = 256;   // score row stride in LDS

DEV __bf16 f2bf(float f) {
  union { float f; unsigned u; } x; x.f = f;
  unsigned r = x.u + 0x7fffu + ((x.u >> 16) & 1u);   // round-to-nearest-even
  unsigned short h = (unsigned short)(r >> 16);
  __bf16 b;
  __builtin_memcpy(&b, &h, sizeof(b));
  return b;
}

DEV v8f vzero8() { v8f z = {0.f,0.f,0.f,0.f,0.f,0.f,0.f,0.f}; return z; }

// A operand (16x32 bf16, row-major source): lanes 0-15 row m, K=k0..k0+7 and
// k0+16..k0+23; lanes 16-31 same rows, K shifted by +8 (ISA 7.12.2 layout).
DEV v16bf load_A_bf16(const __bf16* base, int lda, int k0, int lane) {
  int m  = lane & 15;
  int kb = k0 + ((lane >> 4) << 3);
  const __bf16* p = base + (size_t)m * lda + kb;
  v8bf lo = *(const v8bf*)(p);
  v8bf hi = *(const v8bf*)(p + 16);
  v16bf r;
#pragma unroll
  for (int i = 0; i < 8; ++i) { r[i] = lo[i]; r[i + 8] = hi[i]; }
  return r;
}

// B operand (32x16 bf16): B[k][n] = src[n0+n][k0+k] with src row-major.
// lanes 0-15 hold K=0..7,16..23 of column n; lanes 16-31 hold K=8..15,24..31.
DEV v16bf load_B_rows(const __bf16* base, int ldb, int k0, int n0, int lane) {
  int n  = n0 + (lane & 15);
  int kb = k0 + ((lane >> 4) << 3);
  const __bf16* p = base + (size_t)n * ldb + kb;
  v8bf lo = *(const v8bf*)(p);
  v8bf hi = *(const v8bf*)(p + 16);
  v16bf r;
#pragma unroll
  for (int i = 0; i < 8; ++i) { r[i] = lo[i]; r[i + 8] = hi[i]; }
  return r;
}

DEV v8f wmma_bf16(v16bf a, v16bf b, v8f c) {
  return __builtin_amdgcn_wmma_f32_16x16x32_bf16(false, a, false, b,
                                                 (short)0, c, false, false);
}

// ---------------------------------------------------------------------------
// elementwise: fp32 -> bf16 cast
__global__ void cast_f32_bf16_kernel(const float* __restrict__ src,
                                     __bf16* __restrict__ dst, int n) {
  int i = blockIdx.x * blockDim.x + threadIdx.x;
  if (i < n) dst[i] = f2bf(src[i]);
}

// sinusoid table, rows j in [0,512), pos = j - 255, already bf16.
__global__ void sinusoid_bf16_kernel(__bf16* __restrict__ tbl) {
  int idx = blockIdx.x * blockDim.x + threadIdx.x;  // 512*1024 threads
  int j = idx >> 10;
  int i = idx & 1023;
  float pos = (float)(j - 255);
  float expo = (float)(2 * (i >> 1)) * (1.0f / 1024.0f);
  float inv = __expf(-expo * 9.210340371976184f);   // 1/10000^expo
  float angle = pos * inv;
  float val = (i & 1) ? __cosf(angle) : __sinf(angle);
  tbl[idx] = f2bf(val);
}

// ubk[b,h,z] = sum_d u[h,d] * kR[b,z, h*64+d]
__global__ void ubias_dot_kernel(const float* __restrict__ kR,
                                 const float* __restrict__ u,
                                 float* __restrict__ ubk) {
  int idx = blockIdx.x * blockDim.x + threadIdx.x;  // B*H*S = 65536
  int z = idx & 255;
  int h = (idx >> 8) & 15;
  int b = idx >> 12;
  const float* p  = kR + ((size_t)(b * Ss + z)) * Dd + h * DKk;
  const float* uu = u + h * DKk;
  float s = 0.f;
#pragma unroll 8
  for (int d = 0; d < DKk; ++d) s += uu[d] * p[d];
  ubk[((size_t)(b * Hh + h)) * Ss + z] = s;
}

// vbr[h,j] = sum_d v_bias[h,d] * Pr[j, h*64+d]
__global__ void vbias_dot_kernel(const float* __restrict__ Pr,
                                 const float* __restrict__ vb,
                                 float* __restrict__ vbr) {
  int idx = blockIdx.x * blockDim.x + threadIdx.x;  // H*512 = 8192
  int j = idx & 511;
  int h = idx >> 9;
  const float* p  = Pr + (size_t)j * Dd + h * DKk;
  const float* vv = vb + h * DKk;
  float s = 0.f;
#pragma unroll 8
  for (int d = 0; d < DKk; ++d) s += vv[d] * p[d];
  vbr[(size_t)h * 512 + j] = s;
}

// ---------------------------------------------------------------------------
// C = A @ W^T + bias.  A: (M,K) bf16 row-major, W: (N,K) bf16 row-major.
// MODE 0: fp32 out (M,N); MODE 1: bf16 out (M,N); MODE 2: bf16 out transposed
// (N, ldc) i.e. C^T[n][m].  Block = 4 waves, tile 64(M) x 64(N).
template <int MODE>
__global__ void __launch_bounds__(128)
gemm_bias_kernel(const __bf16* __restrict__ A, const __bf16* __restrict__ W,
                 const float* __restrict__ bias, void* __restrict__ Cout,
                 int M, int N, int K, int ldc) {
  const int lane = threadIdx.x & 31;
  const int wave = threadIdx.x >> 5;
  const int m0 = blockIdx.x * 64 + wave * 16;
  const int n0 = blockIdx.y * 64;
  const __bf16* Abase = A + (size_t)m0 * K;

  v8f acc[4];
#pragma unroll
  for (int t = 0; t < 4; ++t) acc[t] = vzero8();

  for (int k0 = 0; k0 < K; k0 += 32) {
    v16bf a = load_A_bf16(Abase, K, k0, lane);
#pragma unroll
    for (int t = 0; t < 4; ++t) {
      v16bf b = load_B_rows(W, K, k0, n0 + 16 * t, lane);
      acc[t] = wmma_bf16(a, b, acc[t]);
    }
  }

  const int nc = lane & 15, g = lane >> 4;
#pragma unroll
  for (int t = 0; t < 4; ++t) {
    int n = n0 + 16 * t + nc;
    float bv = bias[n];
#pragma unroll
    for (int r = 0; r < 8; ++r) {
      int m = m0 + 8 * g + r;
      float v = acc[t][r] + bv;
      if (MODE == 0)      ((float*)Cout)[(size_t)m * N + n] = v;
      else if (MODE == 1) ((__bf16*)Cout)[(size_t)m * N + n] = f2bf(v);
      else                ((__bf16*)Cout)[(size_t)n * ldc + m] = f2bf(v);
    }
  }
}

// ---------------------------------------------------------------------------
// Attention: one wave per (b, h, 16-row s-tile).  Computes the Toeplitz BD
// strip G = Q @ Pk^T via WMMA, assembles masked scores in LDS, softmax,
// then P @ V via WMMA with V stored transposed for contiguous B loads.
__global__ void __launch_bounds__(32)
relattn_kernel(const __bf16* __restrict__ qb, const __bf16* __restrict__ kb,
               const __bf16* __restrict__ vT, const __bf16* __restrict__ Pkb,
               const float* __restrict__ ubk, const float* __restrict__ vbr,
               __bf16* __restrict__ ctxb) {
  __shared__ float lds[16 * GW + 16 * SSW];
  float* Gw = lds;             // 16 x 272  (BD strip, fp32)
  float* Sw = lds + 16 * GW;   // 16 x 256  (scores / probs, fp32)

  const int lane = threadIdx.x;
  const int st = blockIdx.x & 15;          // s-tile index
  const int h  = (blockIdx.x >> 4) & 15;
  const int b  = blockIdx.x >> 8;
  const int s0 = st * 16;
  const int nzt  = st + 1;                 // causal z-tiles needed
  const int nzt2 = (nzt + 1) & ~1;         // padded even (<= 16)

  const __bf16* qbase  = qb  + ((size_t)(b * Ss + s0)) * Dd + h * DKk;
  const __bf16* kbase  = kb  + ((size_t)(b * Ss)) * Dd + h * DKk;
  const __bf16* pkbase = Pkb + h * DKk;
  const __bf16* vbase  = vT  + ((size_t)(h * DKk)) * (Bb * Ss) + b * Ss;
  const float*  ubk_row = ubk + ((size_t)(b * Hh + h)) * Ss;
  const float*  vbr_row = vbr + (size_t)h * 512;

  const int nc = lane & 15, g = lane >> 4;

  // Q tile operands (16x64 = two K=32 chunks), reused by G and AC passes.
  v16bf a0 = load_A_bf16(qbase, Dd, 0, lane);
  v16bf a1 = load_A_bf16(qbase, Dd, 32, lane);

  // ---- BD strip: G[m][j] = q[s0+m] . Pk[j],  j in [16*(15-st), 272) ----
  for (int jt = 15 - st; jt < 17; ++jt) {
    v8f acc = vzero8();
    acc = wmma_bf16(a0, load_B_rows(pkbase, Dd, 0,  jt * 16, lane), acc);
    acc = wmma_bf16(a1, load_B_rows(pkbase, Dd, 32, jt * 16, lane), acc);
#pragma unroll
    for (int r = 0; r < 8; ++r)
      Gw[(8 * g + r) * GW + jt * 16 + nc] = acc[r];
  }
  __syncthreads();  // single-wave WG: S_NOP + compiler fence

  // ---- scores: AC via WMMA + BD gather + head biases, causal mask ----
  for (int zt = 0; zt < nzt2; ++zt) {
    v8f acc = vzero8();
    acc = wmma_bf16(a0, load_B_rows(kbase, Dd, 0,  zt * 16, lane), acc);
    acc = wmma_bf16(a1, load_B_rows(kbase, Dd, 32, zt * 16, lane), acc);
    int z = zt * 16 + nc;
#pragma unroll
    for (int r = 0; r < 8; ++r) {
      int m = 8 * g + r;
      int s = s0 + m;
      float val = -1.0e30f;
      if (z <= s) {
        int j = z - s + 255;
        val = (acc[r] + Gw[m * GW + j] + ubk_row[z] + vbr_row[j]) * 0.125f;
      }
      Sw[m * SSW + z] = val;
    }
  }
  __syncthreads();

  // ---- softmax over z (2 lanes per row) ----
  {
    int m = lane >> 1, half = lane & 1;
    int ZT = 16 * nzt2, ZH = ZT >> 1;
    float* row = Sw + m * SSW;
    int z0 = half * ZH;
    float mx = -3.0e38f;
    for (int z = z0; z < z0 + ZH; ++z) mx = fmaxf(mx, row[z]);
    mx = fmaxf(mx, __shfl_xor(mx, 1, 32));
    float sum = 0.f;
    for (int z = z0; z < z0 + ZH; ++z) {
      float e = __expf(row[z] - mx);
      row[z] = e;
      sum += e;
    }
    sum += __shfl_xor(sum, 1, 32);
    float inv = 1.0f / sum;
    for (int z = z0; z < z0 + ZH; ++z) row[z] *= inv;
  }
  __syncthreads();

  // ---- ctx = P @ V : A = probs (bf16 from LDS), B = vT (contiguous) ----
  v8f oacc[4];
#pragma unroll
  for (int t = 0; t < 4; ++t) oacc[t] = vzero8();

  for (int kt = 0; kt < (nzt2 >> 1); ++kt) {
    int k0 = kt * 32;
    int m = lane & 15;
    int kbx = k0 + ((lane >> 4) << 3);
    v16bf pa;
#pragma unroll
    for (int i = 0; i < 8; ++i) {
      pa[i]     = f2bf(Sw[m * SSW + kbx + i]);
      pa[i + 8] = f2bf(Sw[m * SSW + kbx + 16 + i]);
    }
#pragma unroll
    for (int t = 0; t < 4; ++t) {
      v16bf bm = load_B_rows(vbase, Bb * Ss, k0, t * 16, lane);
      oacc[t] = wmma_bf16(pa, bm, oacc[t]);
    }
  }

#pragma unroll
  for (int t = 0; t < 4; ++t) {
#pragma unroll
    for (int r = 0; r < 8; ++r) {
      int m = 8 * g + r;
      ctxb[((size_t)(b * Ss + s0 + m)) * Dd + h * DKk + t * 16 + nc] =
          f2bf(oacc[t][r]);
    }
  }
}

// ---------------------------------------------------------------------------
extern "C" void kernel_launch(void* const* d_in, const int* in_sizes, int n_in,
                              void* d_out, int out_size, void* d_ws,
                              size_t ws_size, hipStream_t stream) {
  (void)in_sizes; (void)n_in; (void)out_size; (void)ws_size;
  const float* x  = (const float*)d_in[0];
  const float* Wq = (const float*)d_in[1];
  const float* bq = (const float*)d_in[2];
  const float* Wk = (const float*)d_in[3];
  const float* bk = (const float*)d_in[4];
  const float* Wv = (const float*)d_in[5];
  const float* bv = (const float*)d_in[6];
  const float* Wr = (const float*)d_in[7];
  const float* br = (const float*)d_in[8];
  const float* ub = (const float*)d_in[9];
  const float* vb = (const float*)d_in[10];
  const float* Wo = (const float*)d_in[11];
  const float* bo = (const float*)d_in[12];
  float* out = (float*)d_out;

  char* p = (char*)d_ws;
  auto take = [&](size_t bytes) {
    char* r = p;
    p += (bytes + 255) & ~(size_t)255;
    return (void*)r;
  };
  const size_t Mx = (size_t)Bb * Ss;   // 4096 tokens
  const size_t DW = (size_t)Dd * Dd;   // 1M weights

  __bf16* xb   = (__bf16*)take(Mx * Dd * 2);
  __bf16* wqb  = (__bf16*)take(DW * 2);
  __bf16* wkb  = (__bf16*)take(DW * 2);
  __bf16* wvb  = (__bf16*)take(DW * 2);
  __bf16* wrb  = (__bf16*)take(DW * 2);
  __bf16* wob  = (__bf16*)take(DW * 2);
  __bf16* tblb = (__bf16*)take((size_t)512 * Dd * 2);
  __bf16* qbuf = (__bf16*)take(Mx * Dd * 2);
  __bf16* kbuf = (__bf16*)take(Mx * Dd * 2);
  __bf16* vT   = (__bf16*)take(Mx * Dd * 2);           // transposed (D, B*S)
  float*  kRf  = (float*) take(Mx * Dd * 4);
  __bf16* Pkb  = (__bf16*)take((size_t)512 * Dd * 2);
  float*  Prf  = (float*) take((size_t)512 * Dd * 4);
  float*  ubk  = (float*) take((size_t)Bb * Hh * Ss * 4);
  float*  vbr  = (float*) take((size_t)Hh * 512 * 4);
  __bf16* ctxb = (__bf16*)take(Mx * Dd * 2);

  const int T = 256;
  // casts + sinusoid table
  cast_f32_bf16_kernel<<<(int)(Mx * Dd / T), T, 0, stream>>>(x, xb, (int)(Mx * Dd));
  cast_f32_bf16_kernel<<<(int)(DW / T), T, 0, stream>>>(Wq, wqb, (int)DW);
  cast_f32_bf16_kernel<<<(int)(DW / T), T, 0, stream>>>(Wk, wkb, (int)DW);
  cast_f32_bf16_kernel<<<(int)(DW / T), T, 0, stream>>>(Wv, wvb, (int)DW);
  cast_f32_bf16_kernel<<<(int)(DW / T), T, 0, stream>>>(Wr, wrb, (int)DW);
  cast_f32_bf16_kernel<<<(int)(DW / T), T, 0, stream>>>(Wo, wob, (int)DW);
  sinusoid_bf16_kernel<<<512 * Dd / T, T, 0, stream>>>(tblb);

  // projections (all bf16-WMMA GEMMs)
  dim3 gBig((unsigned)(Mx / 64), Dd / 64);
  dim3 gTbl(512 / 64, Dd / 64);
  gemm_bias_kernel<1><<<gBig, 128, 0, stream>>>(xb, wqb, bq, qbuf, (int)Mx, Dd, Dd, 0);
  gemm_bias_kernel<1><<<gBig, 128, 0, stream>>>(xb, wkb, bk, kbuf, (int)Mx, Dd, Dd, 0);
  gemm_bias_kernel<2><<<gBig, 128, 0, stream>>>(xb, wvb, bv, vT,   (int)Mx, Dd, Dd, (int)Mx);
  gemm_bias_kernel<0><<<gBig, 128, 0, stream>>>(xb, wrb, br, kRf,  (int)Mx, Dd, Dd, 0);
  gemm_bias_kernel<1><<<gTbl, 128, 0, stream>>>(tblb, wkb, bk, Pkb, 512, Dd, Dd, 0);
  gemm_bias_kernel<0><<<gTbl, 128, 0, stream>>>(tblb, wrb, br, Prf, 512, Dd, Dd, 0);

  // rank-1 bias terms
  ubias_dot_kernel<<<Bb * Hh * Ss / T, T, 0, stream>>>(kRf, ub, ubk);
  vbias_dot_kernel<<<Hh * 512 / T, T, 0, stream>>>(Prf, vb, vbr);

  // attention: one wave per (b, h, 16-row s tile)
  relattn_kernel<<<Bb * Hh * (Ss / 16), 32, 0, stream>>>(qbuf, kbuf, vT, Pkb,
                                                         ubk, vbr, ctxb);

  // output projection -> fp32
  gemm_bias_kernel<0><<<gBig, 128, 0, stream>>>(ctxb, wob, bo, out, (int)Mx, Dd, Dd, 0);
}
